// ConvNextBlock_65223373357586
// MI455X (gfx1250) — compile-verified, matrix operand order checked by reading, not run on hardware
//
#include <hip/hip_runtime.h>

// ---------------- problem constants (match reference) ----------------
#define N_VOX   500000
#define CCH     64
#define KOFF    9
#define NPAIR   250000
#define CT_PER_K  (NPAIR / 16)          // 15625 pair-tiles per offset
#define CONV_TILES (KOFF * CT_PER_K)    // 140625
#define MLP_TILES  (N_VOX / 16)         // 31250
#define LEAKK   0.333f

typedef __attribute__((ext_vector_type(16))) __bf16 v16bf;
typedef __attribute__((ext_vector_type(8)))  float  v8f;

union AV { v16bf v; uint4 q[2]; };

__device__ __forceinline__ v8f wmma_bf16(v16bf a, v16bf b, v8f c) {
  // v_wmma_f32_16x16x32_bf16: D = A(16x32) * B(32x16) + C
  return __builtin_amdgcn_wmma_f32_16x16x32_bf16(
      /*neg_a=*/false, a, /*neg_b=*/false, b,
      /*c_mod=*/(short)0, c, /*reuse_a=*/false, /*reuse_b=*/false);
}

__device__ __forceinline__ void atomic_addf(float* p, float v) {
  __hip_atomic_fetch_add(p, v, __ATOMIC_RELAXED, __HIP_MEMORY_SCOPE_AGENT);
}

// ---------------- prep: feat -> bf16, conv_out := b1 broadcast ----------------
__global__ __launch_bounds__(256) void prep_feat_kernel(
    const float* __restrict__ feat, const float* __restrict__ b1,
    __bf16* __restrict__ featbf, float* __restrict__ conv_out) {
  long i = (long)blockIdx.x * 256 + threadIdx.x;   // exactly N_VOX*CCH threads
  featbf[i]   = (__bf16)feat[i];
  conv_out[i] = b1[i & 63];
}

// Pack a (Kdim x Ndim) row-major fp32 weight into bf16 WMMA-B tiles:
// tile = kb*nblocks+nb holds 32 lanes x 16 bf16 contiguous; lane slot j maps to
// K = kb*32 + (j<8?0:16) + (lane<16?0:8) + (j&7), N = nb*16 + (lane&15).
__device__ __forceinline__ void pack_tile(const float* __restrict__ src,
                                          __bf16* __restrict__ dst,
                                          int e, int Ndim, int nblocks) {
  int tile = e >> 9;
  int ln   = (e >> 4) & 31;
  int j    = e & 15;
  int kb = tile / nblocks, nb = tile - kb * nblocks;
  int kk = ((j < 8) ? 0 : 16) + ((ln < 16) ? 0 : 8) + (j & 7);
  int n  = nb * 16 + (ln & 15);
  dst[e] = (__bf16)src[(kb * 32 + kk) * Ndim + n];
}

__global__ __launch_bounds__(256) void prep_weights_kernel(
    const float* __restrict__ W1, const float* __restrict__ W2,
    const float* __restrict__ W3, __bf16* __restrict__ W1B,
    __bf16* __restrict__ W2B, __bf16* __restrict__ W3B,
    float* __restrict__ stats) {
  int tid = blockIdx.x * 256 + threadIdx.x;        // 272*256 = 69632 threads
  if (blockIdx.x == 0 && threadIdx.x < 128) stats[threadIdx.x] = 0.f;
  if (tid < 36864) {                                // W1: 9 x (64x64), 2 kb x 4 nb
    int k = tid >> 12, e = tid & 4095;
    pack_tile(W1 + (k << 12), W1B + (k << 12), e, 64, 4);
  } else if (tid < 53248) {                         // W2: 64x256, 2 kb x 16 nb
    pack_tile(W2, W2B, tid - 36864, 256, 16);
  } else if (tid < 69632) {                         // W3: 256x64, 8 kb x 4 nb
    pack_tile(W3, W3B, tid - 53248, 64, 4);
  }
}

// ---------------- conv1: gather - bf16 WMMA - f32 atomic scatter ----------------
__global__ __launch_bounds__(128) void conv_scatter_kernel(
    const __bf16* __restrict__ featbf, const int* __restrict__ pairs_in,
    const int* __restrict__ pairs_out, const __bf16* __restrict__ W1B,
    float* __restrict__ conv_out) {
  int wave = blockIdx.x * 4 + (threadIdx.x >> 5);
  if (wave >= CONV_TILES) return;                   // wave-uniform: EXEC stays all-1s
  int lane = threadIdx.x & 31;
  int k  = wave / CT_PER_K;
  int tp = wave - k * CT_PER_K;
  int pbase = k * NPAIR + tp * 16;
  int m  = lane & 15;
  int hi = (lane >= 16) ? 8 : 0;

  // Gather A (16 pairs x 32 K) per kb: two b128 loads/lane in WMMA A layout.
  int row = pairs_in[pbase + m];
  const __bf16* fr = featbf + (long)row * CCH;
  AV a[2];
#pragma unroll
  for (int kb = 0; kb < 2; ++kb) {
    int c1 = kb * 32 + hi;
    a[kb].q[0] = *reinterpret_cast<const uint4*>(fr + c1);
    a[kb].q[1] = *reinterpret_cast<const uint4*>(fr + c1 + 16);
  }

  v8f acc[4] = {};
#pragma unroll
  for (int kb = 0; kb < 2; ++kb) {
#pragma unroll
    for (int nb = 0; nb < 4; ++nb) {
      AV b;
      const __bf16* bp = W1B + (((k * 2 + kb) * 4 + nb) << 9) + lane * 16;
      b.q[0] = *reinterpret_cast<const uint4*>(bp);
      b.q[1] = *reinterpret_cast<const uint4*>(bp + 8);
      acc[nb] = wmma_bf16(a[kb].v, b.v, acc[nb]);
    }
  }

  // Scatter: lane holds D rows m = hi + r (r=0..7), cols nb*16 + (lane&15).
  const int4* po = reinterpret_cast<const int4*>(pairs_out + pbase + hi);
  int4 r0 = po[0], r1 = po[1];
  int rows[8] = {r0.x, r0.y, r0.z, r0.w, r1.x, r1.y, r1.z, r1.w};
#pragma unroll
  for (int r = 0; r < 8; ++r) {
    float* op = conv_out + (long)rows[r] * CCH + m;
#pragma unroll
    for (int nb = 0; nb < 4; ++nb) atomic_addf(op + nb * 16, acc[nb][r]);
  }
}

// ---------------- BN stats (sum / sumsq per channel) ----------------
__global__ __launch_bounds__(256) void bn_stats_kernel(
    const float* __restrict__ conv_out, float* __restrict__ stats) {
  __shared__ float ls[256], lq[256];
  int c = threadIdx.x & 63;
  float s = 0.f, sq = 0.f;
  for (long row = blockIdx.x * 4 + (threadIdx.x >> 6); row < N_VOX;
       row += (long)gridDim.x * 4) {
    float v = conv_out[row * CCH + c];
    s += v; sq += v * v;
  }
  ls[threadIdx.x] = s; lq[threadIdx.x] = sq;
  __syncthreads();
  if (threadIdx.x < 64) {
    s  = ls[c] + ls[c + 64] + ls[c + 128] + ls[c + 192];
    sq = lq[c] + lq[c + 64] + lq[c + 128] + lq[c + 192];
    atomic_addf(&stats[c], s);
    atomic_addf(&stats[64 + c], sq);
  }
}

__global__ void bn_finalize_kernel(const float* __restrict__ stats,
                                   const float* __restrict__ gamma,
                                   const float* __restrict__ beta,
                                   float* __restrict__ ss) {
  int c = threadIdx.x;
  if (c < 64) {
    float inv  = 1.f / (float)N_VOX;
    float mean = stats[c] * inv;
    float var  = stats[64 + c] * inv - mean * mean;
    float sc   = gamma[c] * rsqrtf(var + 1e-5f);
    ss[c]      = sc;
    ss[64 + c] = beta[c] - mean * sc;
  }
}

// ---------------- fused BN-apply + MLP(64->256->64) + residual ----------------
__global__ __launch_bounds__(128) void mlp_fused_kernel(
    const float* __restrict__ conv_out, const float* __restrict__ feat,
    const __bf16* __restrict__ W2B, const __bf16* __restrict__ W3B,
    const float* __restrict__ ss, const float* __restrict__ b2,
    const float* __restrict__ b3, float* __restrict__ out) {
  __shared__ __bf16 hbuf[4][16 * 264];   // per-wave h tile, padded 528B row stride
  __shared__ float s_scale[64], s_shift[64], s_b2[256], s_b3[64];
  int tid = threadIdx.x;
  if (tid < 64) { s_scale[tid] = ss[tid]; s_shift[tid] = ss[64 + tid]; s_b3[tid] = b3[tid]; }
  s_b2[tid] = b2[tid]; s_b2[tid + 128] = b2[tid + 128];
  __syncthreads();

  int wv = tid >> 5, lane = tid & 31;
  int t = blockIdx.x * 4 + wv;
  if (t >= MLP_TILES) return;                      // wave-uniform exit
  int m  = lane & 15;
  int hi = (lane >= 16) ? 8 : 0;

  // Build A1 = BN(conv_out rows) as bf16 in WMMA A layout.
  const float* cr = conv_out + ((long)t * 16 + m) * CCH;
  AV a1[2];
#pragma unroll
  for (int kb = 0; kb < 2; ++kb) {
    int cbase = kb * 32 + hi;
#pragma unroll
    for (int h2 = 0; h2 < 2; ++h2) {
#pragma unroll
      for (int j = 0; j < 8; ++j) {
        int col = cbase + h2 * 16 + j;
        float v = cr[col] * s_scale[col] + s_shift[col];
        a1[kb].v[h2 * 8 + j] = (__bf16)v;
      }
    }
  }

  // Stage 1: h(16x256) = A1 * W2, leaky ReLU, spill to LDS (two halves of 128 N).
  __bf16* hb = hbuf[wv];
#pragma unroll
  for (int half = 0; half < 2; ++half) {
    v8f acc[8] = {};
#pragma unroll
    for (int kb = 0; kb < 2; ++kb) {
#pragma unroll
      for (int nbl = 0; nbl < 8; ++nbl) {
        int nb = half * 8 + nbl;
        AV b;
        const __bf16* bp = W2B + ((kb * 16 + nb) << 9) + lane * 16;
        b.q[0] = *reinterpret_cast<const uint4*>(bp);
        b.q[1] = *reinterpret_cast<const uint4*>(bp + 8);
        acc[nbl] = wmma_bf16(a1[kb].v, b.v, acc[nbl]);
      }
    }
#pragma unroll
    for (int nbl = 0; nbl < 8; ++nbl) {
      int n = (half * 8 + nbl) * 16 + m;
      float bias = s_b2[n];
#pragma unroll
      for (int r = 0; r < 8; ++r) {
        float v = acc[nbl][r] + bias;
        v = (v >= 0.f) ? v : LEAKK * v;
        hb[(hi + r) * 264 + n] = (__bf16)v;       // D layout -> row-major LDS
      }
    }
  }
  asm volatile("s_wait_dscnt 0x0" ::: "memory");   // LDS RAW fence (same wave)

  // Stage 2: out(16x64) = h * W3; A2 reloaded from LDS in WMMA A layout.
  v8f acco[4] = {};
#pragma unroll
  for (int kb = 0; kb < 8; ++kb) {
    AV a2;
    const __bf16* hp = hb + m * 264 + kb * 32 + hi;
    a2.q[0] = *reinterpret_cast<const uint4*>(hp);
    a2.q[1] = *reinterpret_cast<const uint4*>(hp + 16);
#pragma unroll
    for (int nb = 0; nb < 4; ++nb) {
      AV b;
      const __bf16* bp = W3B + ((kb * 4 + nb) << 9) + lane * 16;
      b.q[0] = *reinterpret_cast<const uint4*>(bp);
      b.q[1] = *reinterpret_cast<const uint4*>(bp + 8);
      acco[nb] = wmma_bf16(a2.v, b.v, acco[nb]);
    }
  }

  // Epilogue: + b3 + residual, store fp32.
#pragma unroll
  for (int r = 0; r < 8; ++r) {
    long orow = (long)t * 16 + hi + r;
    const float* fr2 = feat + orow * CCH;
    float* op = out + orow * CCH;
#pragma unroll
    for (int nb = 0; nb < 4; ++nb) {
      int n = nb * 16 + m;
      op[n] = acco[nb][r] + s_b3[n] + fr2[n];
    }
  }
}

// ---------------- host-side launch ----------------
extern "C" void kernel_launch(void* const* d_in, const int* in_sizes, int n_in,
                              void* d_out, int out_size, void* d_ws, size_t ws_size,
                              hipStream_t stream) {
  (void)in_sizes; (void)n_in; (void)out_size; (void)ws_size;
  const float* feat      = (const float*)d_in[0];
  const int*   pairs_in  = (const int*)d_in[1];
  const int*   pairs_out = (const int*)d_in[2];
  const float* W1        = (const float*)d_in[3];
  const float* b1        = (const float*)d_in[4];
  const float* gamma     = (const float*)d_in[5];
  const float* beta      = (const float*)d_in[6];
  const float* W2        = (const float*)d_in[7];
  const float* b2        = (const float*)d_in[8];
  const float* W3        = (const float*)d_in[9];
  const float* b3        = (const float*)d_in[10];
  float* out = (float*)d_out;

  // workspace layout (bytes)
  char* ws = (char*)d_ws;
  __bf16* featbf  = (__bf16*)ws;                               //  64,000,000
  float*  convout = (float*)(ws + 64000000);                   // 128,000,000
  __bf16* W1B     = (__bf16*)(ws + 192000000);                 //      73,728
  __bf16* W2B     = (__bf16*)(ws + 192000000 + 73728);         //      32,768
  __bf16* W3B     = (__bf16*)(ws + 192000000 + 73728 + 32768); //      32,768
  float*  stats   = (float*)(ws + 192000000 + 73728 + 65536);  //         512
  float*  ssbuf   = stats + 128;                               //         512

  prep_feat_kernel<<<(N_VOX * CCH) / 256, 256, 0, stream>>>(feat, b1, featbf, convout);
  prep_weights_kernel<<<272, 256, 0, stream>>>(W1, W2, W3, W1B, W2B, W3B, stats);
  conv_scatter_kernel<<<(CONV_TILES + 3) / 4, 128, 0, stream>>>(
      featbf, pairs_in, pairs_out, W1B, convout);
  bn_stats_kernel<<<512, 256, 0, stream>>>(convout, stats);
  bn_finalize_kernel<<<1, 64, 0, stream>>>(stats, gamma, beta, ssbuf);
  mlp_fused_kernel<<<(MLP_TILES + 3) / 4, 128, 0, stream>>>(
      convout, feat, W2B, W3B, ssbuf, b2, b3, out);
}
